// MultiLabelClassifier_76974403879334
// MI455X (gfx1250) — matrix operand order repaired and checked by reading, not compiled, not run on hardware
//
#include <hip/hip_runtime.h>
#include <hip/hip_bf16.h>

typedef __attribute__((ext_vector_type(16))) _Float16 v16h;
typedef __attribute__((ext_vector_type(8)))  _Float16 v8h;
typedef __attribute__((ext_vector_type(8)))  float    v8f;

union FragU { v16h v; v8h h[2]; };

#define WMMA16(a, b, c) __builtin_amdgcn_wmma_f32_16x16x32_f16( \
    false, (a), false, (b), (short)0, (c), false, false)

static constexpr int Bsz = 1024;
static constexpr int Dd  = 512;
static constexpr int Nh  = 16;
static constexpr int Hh  = 512;
static constexpr int Mm  = Bsz * Nh;   // 16384

// ---- CDNA5 async global->LDS copy (ASYNCcnt-tracked, 16B per lane) ----
__device__ __forceinline__ void async_b128(unsigned lds_off, const _Float16* base, int gbyte) {
    asm volatile("global_load_async_to_lds_b128 %0, %1, %2"
                 :: "v"(lds_off), "v"(gbyte), "s"(base) : "memory");
}
__device__ __forceinline__ void wait_async0() {
    asm volatile("s_wait_asynccnt 0x0" ::: "memory");
}
__device__ __forceinline__ unsigned lds_addr(const void* p) {
    return (unsigned)(unsigned long long)p;   // low 32 bits of flat LDS aperture = LDS offset
}
__device__ __forceinline__ float lrelu(float v) { return fmaxf(v, 0.1f * v); }

// ---------------- prep: transpose / fragment-pack + f32 -> f16 ----------------

// xb[(b*16+n)*512 + d] = x[(b*512 + d)*16 + n]
__global__ void pack_xb(const float* __restrict__ x, _Float16* __restrict__ xb) {
    int idx = blockIdx.x * 256 + threadIdx.x;       // Mm*Dd
    int m = idx >> 9, d = idx & 511;
    int b = m >> 4,  n = m & 15;
    xb[idx] = (_Float16)x[(b * Dd + d) * Nh + n];
}

// Fragment-packed B layout (per 512x512 weight, consumed as WMMA B 32x16 frags):
//   PB[((t*16 + c)*32 + L)*16 + i] = W[k*512 + j],  j = t*16 + (L&15), k = c*32 + (L>>4)*16 + i
// -> a wave's B fragment (col tile t, k-chunk c) is 1 KB contiguous, 32B per lane.
__global__ void pack_ws(const float* __restrict__ Ws1, const float* __restrict__ Ws2,
                        _Float16* __restrict__ PB1, _Float16* __restrict__ PB2) {
    int idx = blockIdx.x * 256 + threadIdx.x;       // 2*512*512
    int which = idx >> 18; int r = idx & 262143;
    int i = r & 15, L = (r >> 4) & 31, c = (r >> 9) & 15, t = r >> 13;
    int j = t * 16 + (L & 15);
    int k = c * 32 + (L >> 4) * 16 + i;
    if (which == 0) PB1[r] = (_Float16)Ws1[k * Dd + j];
    else            PB2[r] = (_Float16)Ws2[k * Dd + j];
}

// Same packing per head for Wc1: B[k][j] = Wc1[(n*512 + k)*512 + j]  (j = h)
__global__ void pack_wc(const float* __restrict__ Wc1, _Float16* __restrict__ PBc) {
    int idx = blockIdx.x * 256 + threadIdx.x;       // 16*512*512
    int n = idx >> 18; int r = idx & 262143;
    int i = r & 15, L = (r >> 4) & 31, c = (r >> 9) & 15, t = r >> 13;
    int j = t * 16 + (L & 15);
    int k = c * 32 + (L >> 4) * 16 + i;
    PBc[idx] = (_Float16)Wc1[(n * Dd + k) * Hh + j];
}

// ---------------- shared-layer GEMM: C = f16(lrelu(A @ B + bias)) ----------------
// block tile 128 rows x 256 cols; 8 waves (2x4), 64x64 per wave, 16 WMMA / K-chunk
__global__ __launch_bounds__(256)
void gemm_tile(const _Float16* __restrict__ A, const _Float16* __restrict__ PB,
               const float* __restrict__ bias, _Float16* __restrict__ C) {
    constexpr int KP = 40;                            // 32 + 8 pad (halves)
    __shared__ __align__(16) _Float16 sA[2][128 * KP];

    const int tid  = threadIdx.x;
    const int lane = tid & 31;
    const int wv   = tid >> 5;
    const int wr   = wv >> 2;           // 0..1 (64-row strip)
    const int wc   = wv & 3;            // 0..3 (64-col strip)
    const int lm   = lane & 15;
    const int hl   = lane >> 4;
    const int mBlk = blockIdx.x * 128;
    const int jBlk = blockIdx.y * 256;

    v8f acc[4][4];
    v8f zero = {0.f,0.f,0.f,0.f,0.f,0.f,0.f,0.f};
    for (int i = 0; i < 4; ++i) for (int j = 0; j < 4; ++j) acc[i][j] = zero;

    // loop-invariant packed-B base per nt: chunk delta is a constant immediate offset
    const v8h* pB[4];
    for (int nt = 0; nt < 4; ++nt) {
        int t = (jBlk >> 4) + wc * 4 + nt;
        pB[nt] = reinterpret_cast<const v8h*>(PB + t * 8192 + lane * 16);
    }
    // A fragment LDS bases (per buffer)
    const _Float16* fA[2] = { &sA[0][(wr * 64 + lm) * KP + 16 * hl],
                              &sA[1][(wr * 64 + lm) * KP + 16 * hl] };

    // A staging: 128 rows x 32 halves = 8KB -> 2 async b128 per thread
    const int arow = tid >> 1;
    const int aseg = (tid & 1) * 16;                  // halves
    const unsigned ldsA[2] = { lds_addr(&sA[0][arow * KP + aseg]),
                               lds_addr(&sA[1][arow * KP + aseg]) };
    const int gA = ((mBlk + arow) * 512 + aseg) * 2;  // bytes; +64B per k-chunk

    async_b128(ldsA[0],      A, gA);
    async_b128(ldsA[0] + 16, A, gA + 16);

#pragma unroll
    for (int c = 0; c < 16; ++c) {
        const int buf = c & 1;
        wait_async0();
        __syncthreads();
        if (c < 15) {   // folds at compile time under full unroll
            async_b128(ldsA[buf ^ 1],      A, gA + (c + 1) * 64);
            async_b128(ldsA[buf ^ 1] + 16, A, gA + (c + 1) * 64 + 16);
        }

        FragU bF[4];
        for (int nt = 0; nt < 4; ++nt) {
            bF[nt].h[0] = pB[nt][c * 64];
            bF[nt].h[1] = pB[nt][c * 64 + 1];
        }
        FragU aF[4];
        for (int mt = 0; mt < 4; ++mt) {   // A: K contiguous {0..15}/{16..31}
            const v8h* p = reinterpret_cast<const v8h*>(fA[buf] + mt * 16 * KP);
            aF[mt].h[0] = p[0];
            aF[mt].h[1] = p[1];
        }
        for (int mt = 0; mt < 4; ++mt)
            for (int nt = 0; nt < 4; ++nt)
                acc[mt][nt] = WMMA16(aF[mt].v, bF[nt].v, acc[mt][nt]);
    }

    for (int mt = 0; mt < 4; ++mt)
        for (int nt = 0; nt < 4; ++nt) {
            int j = jBlk + wc * 64 + nt * 16 + lm;
            float bj = bias[j];
            for (int r = 0; r < 8; ++r) {
                int m = mBlk + wr * 64 + mt * 16 + hl * 8 + r;
                C[m * 512 + j] = (_Float16)lrelu(acc[mt][nt][r] + bj);
            }
        }
}

// ---------------- per-head GEMM fused with Wc2 reduction ----------------
// full_logits[m, head] = lrelu(shared_row(m) @ Wc1[head] + bc1[head]) . Wc2[head] + bc2[head]
__global__ __launch_bounds__(256)
void headgemm(const _Float16* __restrict__ SH, const _Float16* __restrict__ PBc,
              const float* __restrict__ bc1, const float* __restrict__ Wc2,
              const float* __restrict__ bc2, float* __restrict__ logits) {
    constexpr int KP = 40;
    __shared__ __align__(16) _Float16 sA[2][64 * KP];
    __shared__ float red[8 * 64];

    const int tid  = threadIdx.x;
    const int lane = tid & 31;
    const int wv   = tid >> 5;          // 64-col strip
    const int lm   = lane & 15;
    const int hl   = lane >> 4;
    const int mBlk = blockIdx.x * 64;
    const int head = blockIdx.y;
    const _Float16* PB = PBc + head * (Dd * Hh);

    v8f acc[4][4];
    v8f zero = {0.f,0.f,0.f,0.f,0.f,0.f,0.f,0.f};
    for (int i = 0; i < 4; ++i) for (int j = 0; j < 4; ++j) acc[i][j] = zero;

    const v8h* pB[4];
    for (int nt = 0; nt < 4; ++nt) {
        int t = wv * 4 + nt;
        pB[nt] = reinterpret_cast<const v8h*>(PB + t * 8192 + lane * 16);
    }
    const _Float16* fA[2] = { &sA[0][lm * KP + 16 * hl],
                              &sA[1][lm * KP + 16 * hl] };

    // A staging: 64 rows x 32 halves = 4KB -> 1 async b128 per thread
    const int arow = tid >> 2;
    const int aseg = (tid & 3) * 8;                   // halves
    const unsigned ldsA[2] = { lds_addr(&sA[0][arow * KP + aseg]),
                               lds_addr(&sA[1][arow * KP + aseg]) };
    const int gA = ((mBlk + arow) * 512 + aseg) * 2;

    async_b128(ldsA[0], SH, gA);

#pragma unroll
    for (int c = 0; c < 16; ++c) {
        const int buf = c & 1;
        wait_async0();
        __syncthreads();
        if (c < 15)
            async_b128(ldsA[buf ^ 1], SH, gA + (c + 1) * 64);

        FragU bF[4];
        for (int nt = 0; nt < 4; ++nt) {
            bF[nt].h[0] = pB[nt][c * 64];
            bF[nt].h[1] = pB[nt][c * 64 + 1];
        }
        FragU aF[4];
        for (int mt = 0; mt < 4; ++mt) {
            const v8h* p = reinterpret_cast<const v8h*>(fA[buf] + mt * 16 * KP);
            aF[mt].h[0] = p[0];
            aF[mt].h[1] = p[1];
        }
        for (int mt = 0; mt < 4; ++mt)
            for (int nt = 0; nt < 4; ++nt)
                acc[mt][nt] = WMMA16(aF[mt].v, bF[nt].v, acc[mt][nt]);
    }

    // fused epilogue: bias + lrelu, dot with Wc2 over this wave's 64 cols
    float w2[4], b1[4];
    for (int nt = 0; nt < 4; ++nt) {
        int j = wv * 64 + nt * 16 + lm;
        w2[nt] = Wc2[head * Hh + j];
        b1[nt] = bc1[head * Hh + j];
    }
    for (int mt = 0; mt < 4; ++mt)
        for (int r = 0; r < 8; ++r) {
            float s = 0.f;
            for (int nt = 0; nt < 4; ++nt)
                s += lrelu(acc[mt][nt][r] + b1[nt]) * w2[nt];
            s += __shfl_xor(s, 1, 32);
            s += __shfl_xor(s, 2, 32);
            s += __shfl_xor(s, 4, 32);
            s += __shfl_xor(s, 8, 32);
            if (lm == 0) red[wv * 64 + mt * 16 + hl * 8 + r] = s;
        }
    __syncthreads();
    if (tid < 64) {
        float s = bc2[head];
        for (int w = 0; w < 8; ++w) s += red[w * 64 + tid];
        logits[(mBlk + tid) * Nh + head] = s;
    }
}

// out[b*16+n] = sigmoid(full_logits[n*B+b, n])
__global__ void finish_out(const float* __restrict__ logits, float* __restrict__ out) {
    int idx = blockIdx.x * 256 + threadIdx.x;  // 16384
    int b = idx >> 4, n = idx & 15;
    float l = logits[(n * Bsz + b) * Nh + n];
    out[idx] = 1.f / (1.f + __expf(-l));
}

extern "C" void kernel_launch(void* const* d_in, const int* in_sizes, int n_in,
                              void* d_out, int out_size, void* d_ws, size_t ws_size,
                              hipStream_t stream) {
    const float* x   = (const float*)d_in[0];
    const float* Ws1 = (const float*)d_in[1];
    const float* bs1 = (const float*)d_in[2];
    const float* Ws2 = (const float*)d_in[3];
    const float* bs2 = (const float*)d_in[4];
    const float* Wc1 = (const float*)d_in[5];
    const float* bc1 = (const float*)d_in[6];
    const float* Wc2 = (const float*)d_in[7];
    const float* bc2 = (const float*)d_in[8];

    float* out         = (float*)d_out;          // (B,1,N) = 16384
    float* full_logits = out + (Bsz * Nh);       // (M,1,N) = 16384*16

    _Float16* ws  = (_Float16*)d_ws;
    _Float16* xb  = ws;                 // 16384*512
    _Float16* h1  = ws + 8388608;       // 16384*512
    _Float16* sh  = ws + 16777216;      // 16384*512
    _Float16* PB1 = ws + 25165824;      // 512*512 fragment-packed
    _Float16* PB2 = ws + 25427968;      // 512*512 fragment-packed
    _Float16* PBc = ws + 25690112;      // 16*512*512 fragment-packed

    pack_xb<<<(Mm * Dd) / 256, 256, 0, stream>>>(x, xb);
    pack_ws<<<(2 * Dd * Dd) / 256, 256, 0, stream>>>(Ws1, Ws2, PB1, PB2);
    pack_wc<<<(Nh * Dd * Hh) / 256, 256, 0, stream>>>(Wc1, PBc);

    gemm_tile<<<dim3(Mm / 128, Dd / 256), 256, 0, stream>>>(xb, PB1, bs1, h1);
    gemm_tile<<<dim3(Mm / 128, Dd / 256), 256, 0, stream>>>(h1, PB2, bs2, sh);

    headgemm<<<dim3(Mm / 64, Nh), 256, 0, stream>>>(sh, PBc, bc1, Wc2, bc2, full_logits);
    finish_out<<<(Bsz * Nh) / 256, 256, 0, stream>>>(full_logits, out);
}
